// MarlinW4A8Linear_54468775248391
// MI455X (gfx1250) — 8-bit matrix operand order repaired and checked by reading, compiled, not run on hardware
//
#include <hip/hip_runtime.h>
#include <hip/hip_bf16.h>


typedef int   v8i __attribute__((ext_vector_type(8)));

#define MDIM   32
#define KDIM   8192
#define NDIM   8192
#define GROUPK 128
#define NGRP   64
#define NTILE  64
#define PITCH  20

__global__ __launch_bounds__(256)
void w4a8_quantize(const float* __restrict__ x,
                   int*   __restrict__ xq32,
                   float* __restrict__ sx,
                   float* __restrict__ rowsum)
{
    __shared__ float red[256];
    __shared__ int   gsum[NGRP];

    const int m = blockIdx.x;
    const int t = threadIdx.x;

    float4 xv[8];
    float amax = 0.0f;
#pragma unroll
    for (int i = 0; i < 8; ++i) {
        xv[i] = *reinterpret_cast<const float4*>(x + (size_t)m * KDIM + 4 * t + 1024 * i);
        amax = fmaxf(amax, fabsf(xv[i].x));
        amax = fmaxf(amax, fabsf(xv[i].y));
        amax = fmaxf(amax, fabsf(xv[i].z));
        amax = fmaxf(amax, fabsf(xv[i].w));
    }
    red[t] = amax;
    __syncthreads();
#pragma unroll
    for (int s = 128; s > 0; s >>= 1) {
        if (t < s) red[t] = fmaxf(red[t], red[t + s]);
        __syncthreads();
    }
    const float mx  = fmaxf(red[0], 1e-8f);
    const float inv = 127.0f / mx;

    if (t < NGRP) gsum[t] = 0;
    __syncthreads();

#pragma unroll
    for (int i = 0; i < 8; ++i) {
        float r0 = fminf(fmaxf(__builtin_rintf(xv[i].x * inv), -127.0f), 127.0f);
        float r1 = fminf(fmaxf(__builtin_rintf(xv[i].y * inv), -127.0f), 127.0f);
        float r2 = fminf(fmaxf(__builtin_rintf(xv[i].z * inv), -127.0f), 127.0f);
        float r3 = fminf(fmaxf(__builtin_rintf(xv[i].w * inv), -127.0f), 127.0f);
        int q0 = (int)r0, q1 = (int)r1, q2 = (int)r2, q3 = (int)r3;
        int packed = (q0 & 0xFF) | ((q1 & 0xFF) << 8) | ((q2 & 0xFF) << 16) | ((q3 & 0xFF) << 24);
        xq32[m * (KDIM / 4) + t + 256 * i] = packed;
        atomicAdd(&gsum[(t >> 5) + 8 * i], q0 + q1 + q2 + q3);
    }
    __syncthreads();
    if (t < NGRP) rowsum[m * NGRP + t] = (float)gsum[t];
    if (t == 0)   sx[m] = mx / 127.0f;
}

__global__ __launch_bounds__(128)
void w4a8_gemm(const int*   __restrict__ qw,
               const float* __restrict__ scales,
               const int*   __restrict__ qzeros,
               const float* __restrict__ bias,
               const signed char* __restrict__ xq,
               const float* __restrict__ sx,
               const float* __restrict__ rowsum,
               float* __restrict__ out)
{
    __shared__ int lds[NTILE * PITCH];

    const int t    = threadIdx.x;
    const int n0   = blockIdx.x * NTILE;
    const int wv   = t >> 5;
    const int lane = t & 31;
    const int l15  = lane & 15;
    const int hi   = lane >> 4;
    const int nloc = wv * 16 + l15;
    const int n    = n0 + nloc;

    const int s_n    = t & 63;
    const int s_half = t >> 6;

    float facc[2][8];
#pragma unroll
    for (int mt = 0; mt < 2; ++mt)
#pragma unroll
        for (int r = 0; r < 8; ++r) facc[mt][r] = 0.0f;

    for (int g = 0; g < NGRP; ++g) {
        v8i iacc[2];
        iacc[0] = (v8i){0, 0, 0, 0, 0, 0, 0, 0};
        iacc[1] = (v8i){0, 0, 0, 0, 0, 0, 0, 0};

#pragma unroll 1
        for (int c = 0; c < 2; ++c) {
            const int kb = g * GROUPK + c * 64;

#pragma unroll
            for (int i = 0; i < 8; ++i) {
                const int k4 = s_half * 8 + i;
                const int* src = qw + (size_t)(kb + 4 * k4) * NDIM + n0 + s_n;
                const int q0 = src[0 * NDIM];
                const int q1 = src[1 * NDIM];
                const int q2 = src[2 * NDIM];
                const int q3 = src[3 * NDIM];
                lds[s_n * PITCH + k4] = q0 | (q1 << 8) | (q2 << 16) | (q3 << 24);
            }
            if (kb + 64 < KDIM) {
                __builtin_prefetch(qw + (size_t)(kb + 64) * NDIM + n0 + s_n, 0, 1);
            }
            __syncthreads();

            const int4 b0 = *reinterpret_cast<const int4*>(&lds[nloc * PITCH + hi * 4]);
            const int4 b1 = *reinterpret_cast<const int4*>(&lds[nloc * PITCH + 8 + hi * 4]);
            v8i bfrag;
            bfrag[0] = b0.x; bfrag[1] = b0.y; bfrag[2] = b0.z; bfrag[3] = b0.w;
            bfrag[4] = b1.x; bfrag[5] = b1.y; bfrag[6] = b1.z; bfrag[7] = b1.w;

#pragma unroll
            for (int mt = 0; mt < 2; ++mt) {
                const int row = mt * 16 + l15;
                const int2* ap = reinterpret_cast<const int2*>(
                    xq + (size_t)row * KDIM + kb + hi * 16);
                v8i afrag;
#pragma unroll
                for (int p = 0; p < 4; ++p) {
                    const int2 dv = ap[(p & 1) + (p >> 1) * 4];
                    afrag[2 * p]     = dv.x;
                    afrag[2 * p + 1] = dv.y;
                }
                iacc[mt] = __builtin_amdgcn_wmma_i32_16x16x64_iu8(
                    true, afrag, false, bfrag, iacc[mt], false, false);
            }
            __syncthreads();
        }

        const float sc = scales[(size_t)g * NDIM + n];
        const float zp = (float)qzeros[(size_t)g * NDIM + n];
#pragma unroll
        for (int mt = 0; mt < 2; ++mt) {
#pragma unroll
            for (int r = 0; r < 8; ++r) {
                const int row = mt * 16 + hi * 8 + r;
                const float rs = rowsum[row * NGRP + g];
                facc[mt][r] += sc * ((float)iacc[mt][r] - zp * rs);
            }
        }
    }

    const float bv = bias[n];
#pragma unroll
    for (int mt = 0; mt < 2; ++mt) {
#pragma unroll
        for (int r = 0; r < 8; ++r) {
            const int row = mt * 16 + hi * 8 + r;
            out[(size_t)row * NDIM + n] = sx[row] * facc[mt][r] + bv;
        }
    }
}

extern "C" void kernel_launch(void* const* d_in, const int* in_sizes, int n_in,
                              void* d_out, int out_size, void* d_ws, size_t ws_size,
                              hipStream_t stream)
{
    const float* x      = (const float*)d_in[0];
    const int*   qw     = (const int*)d_in[1];
    const float* scales = (const float*)d_in[2];
    const int*   qzeros = (const int*)d_in[3];
    const float* bias   = (const float*)d_in[4];
    float*       out    = (float*)d_out;

    char* ws = (char*)d_ws;
    int*   xq32   = (int*)ws;
    float* sx     = (float*)(ws + 262144);
    float* rowsum = (float*)(ws + 262144 + 128);

    w4a8_quantize<<<MDIM, 256, 0, stream>>>(x, xq32, sx, rowsum);

    w4a8_gemm<<<NDIM / NTILE, 128, 0, stream>>>(
        qw, scales, qzeros, bias,
        (const signed char*)xq32, sx, rowsum, out);
}
